// MultiHeadAttention_39109972198359
// MI455X (gfx1250) — matrix operand order repaired and checked by reading, not compiled, not run on hardware
//
#include <hip/hip_runtime.h>
#include <hip/hip_bf16.h>
#include <math.h>

// ---------------------------------------------------------------------------
// MI455X (gfx1250) multi-head attention:
//   bf16 WMMA 16x16x32, fp32 accumulate, flash softmax (log2 domain),
//   async-to-LDS (b128) double-buffered staging (ASYNCcnt),
//   diagonal-only causal mask, batched cross-row bpermute reductions.
//   All LDS fragment reads are per-lane contiguous (ds_load_b128).
// ---------------------------------------------------------------------------

typedef __bf16 bf16_t;
typedef __attribute__((ext_vector_type(16))) __bf16 v16bf;
typedef __attribute__((ext_vector_type(8)))  float  v8f;

typedef int v2i_t __attribute__((ext_vector_type(2)));
typedef int v4i_t __attribute__((ext_vector_type(4)));
typedef __attribute__((address_space(1))) v2i_t gv2i;
typedef __attribute__((address_space(3))) v2i_t sv2i;
typedef __attribute__((address_space(1))) v4i_t gv4i;
typedef __attribute__((address_space(3))) v4i_t sv4i;

constexpr int B_ = 8, S_ = 1024, D_ = 1024, H_ = 16, DK_ = 64;
constexpr int NQKV = B_ * S_ * D_;   // 8,388,608
constexpr int NW   = D_ * D_;        // 1,048,576

// ---- CDNA5 async global->LDS copy (ASYNCcnt), 16-byte granule --------------
__device__ inline void async_cp16(bf16_t* lds, const bf16_t* g) {
#if __has_builtin(__builtin_amdgcn_global_load_async_to_lds_b128)
  __builtin_amdgcn_global_load_async_to_lds_b128((gv4i*)g, (sv4i*)lds, 0, 0);
#elif __has_builtin(__builtin_amdgcn_global_load_async_to_lds_b64)
  __builtin_amdgcn_global_load_async_to_lds_b64((gv2i*)g, (sv2i*)lds, 0, 0);
  __builtin_amdgcn_global_load_async_to_lds_b64((gv2i*)(g + 4), (sv2i*)(lds + 4), 0, 0);
#else
  *(unsigned long long*)lds       = *(const unsigned long long*)g;
  *(unsigned long long*)(lds + 4) = *(const unsigned long long*)(g + 4);
#endif
}

__device__ inline void wait_async0() {
#if __has_builtin(__builtin_amdgcn_s_wait_asynccnt)
  __builtin_amdgcn_s_wait_asynccnt(0);
#else
  asm volatile("s_wait_asynccnt 0" ::: "memory");
#endif
}

__device__ inline float fast_exp2(float x) {
#if __has_builtin(__builtin_amdgcn_exp2f)
  return __builtin_amdgcn_exp2f(x);     // raw v_exp_f32
#else
  return exp2f(x);
#endif
}

__device__ inline float fast_rcp(float x) {
#if __has_builtin(__builtin_amdgcn_rcpf)
  return __builtin_amdgcn_rcpf(x);      // raw v_rcp_f32
#else
  return 1.0f / x;
#endif
}

__device__ inline v8f zero8() {
  v8f z;
#pragma unroll
  for (int i = 0; i < 8; ++i) z[i] = 0.f;
  return z;
}

__device__ inline v8f wmma_bf16(v16bf a, v16bf b, v8f c) {
  return __builtin_amdgcn_wmma_f32_16x16x32_bf16(false, a, false, b,
                                                 (short)0, c, false, false);
}

// A fragment: 16(M) x 32(K) bf16, row-major, leading dim `ld`.
// Per-lane contiguous 8+8 halfwords -> 2x ds_load_b128.
__device__ inline v16bf load_a_frag(const bf16_t* base, int ld, int lane) {
  const int m  = lane & 15;
  const int kh = (lane >> 4) & 1;
  const bf16_t* r = base + (size_t)m * ld;
  v16bf a;
#pragma unroll
  for (int i = 0; i < 8; ++i) a[i] = r[kh * 16 + i];
#pragma unroll
  for (int i = 0; i < 8; ++i) a[8 + i] = r[kh * 16 + 8 + i];
  return a;
}

// B fragment from an N-major (transposed) tile: B[kk][n] = src[n*ld + kk].
// Per-lane contiguous 16 halfwords -> 2x ds_load_b128.
__device__ inline v16bf load_bT_frag(const bf16_t* src, int ld, int lane) {
  const int n  = lane & 15;
  const int kh = (lane >> 4) & 1;
  const bf16_t* r = src + (size_t)n * ld + kh * 16;
  v16bf b;
#pragma unroll
  for (int i = 0; i < 16; ++i) b[i] = r[i];
  return b;
}

// ---------------------------------------------------------------------------
__global__ __launch_bounds__(256)
void cast_f32_bf16(const float* __restrict__ x, bf16_t* __restrict__ y, int n) {
  int i0 = blockIdx.x * 1024 + threadIdx.x;
#pragma unroll
  for (int j = 0; j < 4; ++j) {
    int i = i0 + j * 256;
    if (i < n) y[i] = (bf16_t)x[i];
  }
}

// Transpose-cast: w[R][C] (f32) -> wt[C][R] (bf16). 32x32 tiles via LDS.
__global__ __launch_bounds__(256)
void cast_transpose_f32_bf16(const float* __restrict__ w, bf16_t* __restrict__ wt,
                             int R, int C) {
  __shared__ float tile[32][33];
  const int r0 = blockIdx.y * 32, c0 = blockIdx.x * 32;
  const int tx = threadIdx.x & 31, ty = threadIdx.x >> 5;   // 32 x 8
#pragma unroll
  for (int j = 0; j < 32; j += 8)
    tile[ty + j][tx] = w[(size_t)(r0 + ty + j) * C + c0 + tx];
  __syncthreads();
#pragma unroll
  for (int j = 0; j < 32; j += 8)
    wt[(size_t)(c0 + ty + j) * R + r0 + tx] = (bf16_t)tile[tx][ty + j];
}

// ---------------------------------------------------------------------------
// C[M,N] = A[M,K] @ W[K,N] with W given TRANSPOSED (WT[N][K]).
// 256 threads = 8 waves arranged 4(M) x 2(N); 128x128 block tile; 64-wide
// K steps, double-buffered async-b128 staging. Each wave owns a 32x64
// quadrant: 16 WMMAs per K-step from only 4 A-frags + 8 B-frags (each B
// fragment feeds two back-to-back WMMAs).
// MODE: 0 = bf16 row-major out, 1 = f32 row-major out,
//       2 = bf16 out transposed per head: Vt[(bh*DK+dk)*S + s2]
template <int MODE>
__global__ __launch_bounds__(256)
void gemm_wmma(const bf16_t* __restrict__ A, const bf16_t* __restrict__ WT,
               bf16_t* __restrict__ Cb, float* __restrict__ Cf,
               int Md, int Nd, int Kd) {
  __shared__ bf16_t As[2][128][72];     // 128x64 A tile (+8 pad; 144B rows)
  __shared__ bf16_t BsT[2][128][72];    // 128(n) x 64(k) WT tile
  const int tid  = threadIdx.x;
  const int lane = tid & 31;
  const int wid  = tid >> 5;
  const int wm   = wid >> 1;            // 0..3: 32-row M quadrant
  const int wn   = wid & 1;             // 0..1: 64-col N half
  const int m0   = blockIdx.y * 128;
  const int n0   = blockIdx.x * 128;

  auto stage = [&](int k0, int bi) {
#pragma unroll
    for (int j = 0; j < 4; ++j) {       // A: 128 rows x 8 chunks
      int idx = tid + j * 256;
      int r = idx >> 3, c = idx & 7;
      async_cp16(&As[bi][r][c * 8], A + (size_t)(m0 + r) * Kd + k0 + c * 8);
    }
#pragma unroll
    for (int j = 0; j < 4; ++j) {       // WT: 128 n-rows x 8 chunks
      int idx = tid + j * 256;
      int r = idx >> 3, c = idx & 7;
      async_cp16(&BsT[bi][r][c * 8], WT + (size_t)(n0 + r) * Kd + k0 + c * 8);
    }
  };

  v8f acc[2][4];
#pragma unroll
  for (int mh = 0; mh < 2; ++mh)
#pragma unroll
    for (int t = 0; t < 4; ++t) acc[mh][t] = zero8();

  const int nk = Kd / 64;
  stage(0, 0);
  for (int ik = 0; ik < nk; ++ik) {
    wait_async0();
    __syncthreads();
    if (ik + 1 < nk) stage((ik + 1) * 64, (ik + 1) & 1);   // overlap
    const int bi = ik & 1;

#pragma unroll
    for (int kh2 = 0; kh2 < 2; ++kh2) {
      v16bf a0 = load_a_frag(&As[bi][wm * 32][kh2 * 32],      72, lane);
      v16bf a1 = load_a_frag(&As[bi][wm * 32 + 16][kh2 * 32], 72, lane);
#pragma unroll
      for (int t = 0; t < 4; ++t) {
        v16bf b = load_bT_frag(&BsT[bi][wn * 64 + t * 16][kh2 * 32], 72, lane);
        acc[0][t] = wmma_bf16(a0, b, acc[0][t]);
        acc[1][t] = wmma_bf16(a1, b, acc[1][t]);
      }
    }
  }

  const int hf = lane >> 4, nn = lane & 15;
#pragma unroll
  for (int mh = 0; mh < 2; ++mh)
#pragma unroll
    for (int t = 0; t < 4; ++t)
#pragma unroll
      for (int r = 0; r < 8; ++r) {
        const int m = m0 + wm * 32 + mh * 16 + r + 8 * hf;
        const int n = n0 + wn * 64 + t * 16 + nn;
        const float val = acc[mh][t][r];
        if (MODE == 1) {
          Cf[(size_t)m * Nd + n] = val;
        } else if (MODE == 0) {
          Cb[(size_t)m * Nd + n] = (bf16_t)val;
        } else {
          // flat .view head split of row-major [B,S,D]:
          const int b  = m >> 10;             // S_ = 1024
          const int so = m & 1023;
          const int fl = so * D_ + n;         // < 2^20
          const int h  = fl >> 16;            // / (S_*DK_)
          const int s2 = (fl >> 6) & 1023;
          const int dk = n & 63;
          Cb[((size_t)(b * H_ + h) * DK_ + dk) * S_ + s2] = (bf16_t)val;
        }
      }
}

// ---------------------------------------------------------------------------
// Flash attention over one (b,h). Q,K: contiguous [S,DK] slabs; V given
// TRANSPOSED per head: Vt[b,h][dk][s]. Block = 4 waves; wave w handles
// q-tile blockIdx.x*4+w. 64-key blocks, async-b128 double buffering; only
// the final (diagonal) block applies the causal mask (wave-uniform).
// Softmax runs in the log2 domain: Q prescaled by log2(e)/sqrt(DK).
__global__ __launch_bounds__(128)
void attn_wmma(const bf16_t* __restrict__ Q, const bf16_t* __restrict__ K,
               const bf16_t* __restrict__ Vt, bf16_t* __restrict__ O) {
  __shared__ bf16_t Ks[2][64][72];      // [key][dk]
  __shared__ bf16_t VsT[2][64][72];     // [dk][key]
  __shared__ bf16_t Ps[4][16][72];      // per-wave P tile [qrow][key]

  const int tid  = threadIdx.x;
  const int lane = tid & 31;
  const int wid  = tid >> 5;
  const int bh   = blockIdx.y;
  const int qt   = blockIdx.x * 4 + wid;
  const size_t base = (size_t)bh * S_ * DK_;
  const bf16_t* Qh  = Q  + base;
  const bf16_t* Kh  = K  + base;
  const bf16_t* Vth = Vt + base;        // [DK_][S_] slab
  bf16_t*       Oh  = O  + base;
  const int hf = lane >> 4, nn = lane & 15;

  auto stageKV = [&](int kbase, int bi) {
#pragma unroll
    for (int j = 0; j < 4; ++j) {
      int idx = tid + j * 128;
      int r = idx >> 3, c = idx & 7;
      async_cp16(&Ks[bi][r][c * 8],  Kh  + (size_t)(kbase + r) * DK_ + c * 8);
      async_cp16(&VsT[bi][r][c * 8], Vth + (size_t)r * S_ + kbase + c * 8);
    }
  };

  // Q fragments in registers; fold log2(e)/sqrt(DK) into Q so softmax can use
  // raw v_exp_f32 (exp2) with no per-element ln2 conversion multiplies.
  constexpr float QSCALE = 0.125f * 1.44269504088896340736f;
  v16bf aQ0 = load_a_frag(Qh + (size_t)qt * 16 * DK_, DK_, lane);
  v16bf aQ1 = load_a_frag(Qh + (size_t)qt * 16 * DK_ + 32, DK_, lane);
#pragma unroll
  for (int i = 0; i < 16; ++i) {
    aQ0[i] = (bf16_t)((float)aQ0[i] * QSCALE);
    aQ1[i] = (bf16_t)((float)aQ1[i] * QSCALE);
  }

  v8f o[4];
#pragma unroll
  for (int t = 0; t < 4; ++t) o[t] = zero8();
  float mrow[8], lrow[8];
#pragma unroll
  for (int r = 0; r < 8; ++r) { mrow[r] = -INFINITY; lrow[r] = 0.f; }

  const int nkb = blockIdx.x + 1;       // causal: 64-key blocks 0..diag
  stageKV(0, 0);
  for (int kb = 0; kb < nkb; ++kb) {
    const int kbase = kb * 64;
    wait_async0();
    __syncthreads();
    if (kb + 1 < nkb) stageKV((kb + 1) * 64, (kb + 1) & 1);
    const int bi = kb & 1;
    const bool diag = (kb == nkb - 1);  // wave-uniform

    // scores (log2 domain): four 16x16 tiles over 64 keys (2 WMMAs each)
    v8f s[4];
#pragma unroll
    for (int j = 0; j < 4; ++j) {
      s[j] = wmma_bf16(aQ0, load_bT_frag(&Ks[bi][j * 16][0],  72, lane), zero8());
      s[j] = wmma_bf16(aQ1, load_bT_frag(&Ks[bi][j * 16][32], 72, lane), s[j]);
    }

    if (diag) {                          // causal mask, in place
#pragma unroll
      for (int j = 0; j < 4; ++j)
#pragma unroll
        for (int r = 0; r < 8; ++r)
          if (kbase + j * 16 + nn > qt * 16 + r + 8 * hf) s[j][r] = -INFINITY;
    }

    // Row maxima; reductions batched across the 8 independent rows so the
    // ds_bpermute latencies overlap (8 in flight per butterfly step).
    float mx[8];
#pragma unroll
    for (int r = 0; r < 8; ++r)
      mx[r] = fmaxf(fmaxf(s[0][r], s[1][r]), fmaxf(s[2][r], s[3][r]));
#pragma unroll
    for (int d = 1; d <= 8; d <<= 1)
#pragma unroll
      for (int r = 0; r < 8; ++r)
        mx[r] = fmaxf(mx[r], __shfl_xor(mx[r], d, 32));

    float corr[8], sum[8];
#pragma unroll
    for (int r = 0; r < 8; ++r) {
      const float mnew = fmaxf(mrow[r], mx[r]);
      corr[r] = fast_exp2(mrow[r] - mnew);
      mrow[r] = mnew;
      float t0 = 0.f;
#pragma unroll
      for (int j = 0; j < 4; ++j) {
        s[j][r] = fast_exp2(s[j][r] - mnew);
        t0 += s[j][r];
      }
      sum[r] = t0;
    }
#pragma unroll
    for (int d = 1; d <= 8; d <<= 1)
#pragma unroll
      for (int r = 0; r < 8; ++r)
        sum[r] += __shfl_xor(sum[r], d, 32);

#pragma unroll
    for (int r = 0; r < 8; ++r) {
      lrow[r] = lrow[r] * corr[r] + sum[r];
#pragma unroll
      for (int j = 0; j < 4; ++j)
        Ps[wid][r + 8 * hf][j * 16 + nn] = (bf16_t)s[j][r];
#pragma unroll
      for (int t = 0; t < 4; ++t) o[t][r] *= corr[r];
    }

    // O += P(16x64) @ V(64x64): 8 WMMAs, B-frags from [dk][key] tile
    v16bf aP0 = load_a_frag(&Ps[wid][0][0],  72, lane);
    v16bf aP1 = load_a_frag(&Ps[wid][0][32], 72, lane);
#pragma unroll
    for (int t = 0; t < 4; ++t) {
      o[t] = wmma_bf16(aP0, load_bT_frag(&VsT[bi][t * 16][0],  72, lane), o[t]);
      o[t] = wmma_bf16(aP1, load_bT_frag(&VsT[bi][t * 16][32], 72, lane), o[t]);
    }
  }

#pragma unroll
  for (int r = 0; r < 8; ++r) {
    const float inv = fast_rcp(lrow[r]);
#pragma unroll
    for (int t = 0; t < 4; ++t)
      Oh[(size_t)(qt * 16 + r + 8 * hf) * DK_ + t * 16 + nn] =
          (bf16_t)(o[t][r] * inv);
  }
}

// ---------------------------------------------------------------------------
extern "C" void kernel_launch(void* const* d_in, const int* in_sizes, int n_in,
                              void* d_out, int out_size, void* d_ws, size_t ws_size,
                              hipStream_t stream) {
  (void)in_sizes; (void)n_in; (void)out_size; (void)ws_size;
  const float* q  = (const float*)d_in[0];
  const float* k  = (const float*)d_in[1];
  const float* v  = (const float*)d_in[2];
  // d_in[3] = causal tril mask; reproduced analytically (k_idx <= q_idx).
  const float* wq = (const float*)d_in[4];
  const float* wk = (const float*)d_in[5];
  const float* wv = (const float*)d_in[6];
  const float* wo = (const float*)d_in[7];

  char* p = (char*)d_ws;
  auto take = [&](size_t bytes) {
    char* r = p;
    p += (bytes + 255) & ~(size_t)255;
    return r;
  };
  bf16_t* qb   = (bf16_t*)take((size_t)NQKV * 2);
  bf16_t* kb2  = (bf16_t*)take((size_t)NQKV * 2);
  bf16_t* vb   = (bf16_t*)take((size_t)NQKV * 2);
  bf16_t* wqt  = (bf16_t*)take((size_t)NW * 2);   // transposed weights
  bf16_t* wkt  = (bf16_t*)take((size_t)NW * 2);
  bf16_t* wvt  = (bf16_t*)take((size_t)NW * 2);
  bf16_t* wot  = (bf16_t*)take((size_t)NW * 2);
  bf16_t* Qp   = (bf16_t*)take((size_t)NQKV * 2);
  bf16_t* Kp   = (bf16_t*)take((size_t)NQKV * 2);
  bf16_t* Vtp  = (bf16_t*)take((size_t)NQKV * 2); // V projected, [b,h][dk][s]
  bf16_t* AO   = (bf16_t*)take((size_t)NQKV * 2);

  cast_f32_bf16<<<dim3((NQKV + 1023) / 1024), dim3(256), 0, stream>>>(q, qb,  NQKV);
  cast_f32_bf16<<<dim3((NQKV + 1023) / 1024), dim3(256), 0, stream>>>(k, kb2, NQKV);
  cast_f32_bf16<<<dim3((NQKV + 1023) / 1024), dim3(256), 0, stream>>>(v, vb,  NQKV);

  dim3 tg(D_ / 32, D_ / 32);
  cast_transpose_f32_bf16<<<tg, 256, 0, stream>>>(wq, wqt, D_, D_);
  cast_transpose_f32_bf16<<<tg, 256, 0, stream>>>(wk, wkt, D_, D_);
  cast_transpose_f32_bf16<<<tg, 256, 0, stream>>>(wv, wvt, D_, D_);
  cast_transpose_f32_bf16<<<tg, 256, 0, stream>>>(wo, wot, D_, D_);

  // projections: [8192x1024] @ [1024x1024]
  dim3 gg(D_ / 128, (B_ * S_) / 128);  // (8, 64)
  gemm_wmma<0><<<gg, 256, 0, stream>>>(qb,  wqt, Qp,  nullptr, B_ * S_, D_, D_);
  gemm_wmma<0><<<gg, 256, 0, stream>>>(kb2, wkt, Kp,  nullptr, B_ * S_, D_, D_);
  gemm_wmma<2><<<gg, 256, 0, stream>>>(vb,  wvt, Vtp, nullptr, B_ * S_, D_, D_);

  // flash attention: 16 q-tile groups x (B*H = 128) heads
  attn_wmma<<<dim3((S_ / 16) / 4, B_ * H_), 128, 0, stream>>>(Qp, Kp, Vtp, AO);

  // output projection -> fp32 d_out
  gemm_wmma<1><<<gg, 256, 0, stream>>>(AO, wot, nullptr, (float*)d_out,
                                       B_ * S_, D_, D_);
}
